// DAttention_75453985457398
// MI455X (gfx1250) — matrix operand order repaired and checked by reading, not compiled, not run on hardware
//
#include <hip/hip_runtime.h>
#include <hip/hip_bf16.h>

typedef __attribute__((ext_vector_type(16))) __bf16       v16bf;
typedef __attribute__((ext_vector_type(8)))  float        v8f;
typedef __attribute__((ext_vector_type(4)))  unsigned int u32x4;

#define B_    8
#define C_    384
#define H_    40
#define W_    40
#define N_    1600      // H*W
#define NH_   8
#define NG_   4
#define GC_   96        // C/NG
#define HC_   48        // C/NH
#define HCP_  64        // padded head channels for WMMA K-dim
#define SCALE_ 0.14433756729740643f  // 48^-0.5

// ---------------------------------------------------------------------------
// Fragment helpers: a v16bf WMMA operand = two 16B loads per lane.
// A-frag (16x32, MxK): lane m=l%16, g=l/16: K = 16g + {0..15}
// B-frag (32x16, KxN): lane n=l%16, g=l/16: K = 16g + {0..15}
// ---------------------------------------------------------------------------
union FragU { u32x4 q[2]; v16bf v; };

__device__ __forceinline__ v16bf load_frag(const __bf16* p0, const __bf16* p1) {
  FragU u;
  u.q[0] = *reinterpret_cast<const u32x4*>(p0);
  u.q[1] = *reinterpret_cast<const u32x4*>(p1);
  return u.v;
}

#define WMMA_BF16(A, Bm, Cacc) \
  __builtin_amdgcn_wmma_f32_16x16x32_bf16(false, (A), false, (Bm), (short)0, (Cacc), false, false)

__device__ __forceinline__ float wave_sum32(float v) {
  #pragma unroll
  for (int m = 1; m < 32; m <<= 1) v += __shfl_xor(v, m, 32);
  return v;
}

// ---------------------------------------------------------------------------
// fp32 -> bf16 flat convert (weights)
// ---------------------------------------------------------------------------
__global__ void k_cvt_bf16(const float* __restrict__ src, __bf16* __restrict__ dst, int n) {
  int i = blockIdx.x * 256 + threadIdx.x;
  if (i < n) dst[i] = (__bf16)src[i];
}

// ---------------------------------------------------------------------------
// x (B,C,n) fp32 -> xb (B,n,C) bf16  (LDS-tiled transpose)
// ---------------------------------------------------------------------------
__global__ void k_transpose_x(const float* __restrict__ x, __bf16* __restrict__ xb) {
  __shared__ float t[32][33];
  int p0 = blockIdx.x * 32, c0 = blockIdx.y * 32, b = blockIdx.z;
  int tx = threadIdx.x, ty = threadIdx.y;
  #pragma unroll
  for (int i = ty; i < 32; i += 8)
    t[i][tx] = x[((size_t)b * C_ + c0 + i) * N_ + p0 + tx];
  __syncthreads();
  #pragma unroll
  for (int i = ty; i < 32; i += 8)
    xb[((size_t)b * N_ + p0 + i) * C_ + c0 + tx] = (__bf16)t[tx][i];
}

// ---------------------------------------------------------------------------
// Generic 384x384 x (b,1600) WMMA GEMM:  y[b][o][p] = sum_i W[o][i]*Xt[b][p][i] + bias[o]
// mode 0: Q  -> qf32 (B,C,n) fp32  AND qp (B*NH, n, 64) bf16 * SCALE (pad 48..63 zeroed)
// mode 1: K  -> kp (B*NH, n, 64) bf16 (pad zeroed)
// mode 2: V  -> vrow (B*NH, 48, n) bf16
// mode 3: O  -> d_out (B,C,n) fp32
// ---------------------------------------------------------------------------
__global__ void k_gemm384(const __bf16* __restrict__ Wb, const __bf16* __restrict__ Xt,
                          const float* __restrict__ bias, int mode,
                          float* __restrict__ outF, __bf16* __restrict__ outPad,
                          __bf16* __restrict__ outRow) {
  int lane = threadIdx.x & 31;
  int w    = threadIdx.x >> 5;
  int ln   = lane & 15, g = lane >> 4;
  int p0 = (blockIdx.x * 4 + w) * 16;
  int o0 = blockIdx.y * 16;
  int b  = blockIdx.z;

  const __bf16* arow = Wb + (size_t)(o0 + ln) * C_;
  const __bf16* brow = Xt + ((size_t)b * N_ + p0 + ln) * C_;

  v8f acc = {};
  #pragma unroll
  for (int i0 = 0; i0 < C_; i0 += 32) {
    v16bf af = load_frag(arow + i0 + 16 * g, arow + i0 + 16 * g + 8);
    v16bf bf = load_frag(brow + i0 + 16 * g, brow + i0 + 16 * g + 8);
    acc = WMMA_BF16(af, bf, acc);
  }

  #pragma unroll
  for (int r = 0; r < 8; ++r) {
    int o = o0 + r + 8 * g;
    int p = p0 + ln;
    float val = acc[r] + bias[o];
    int h = o / HC_, c = o % HC_;
    if (mode == 0) {
      outF[((size_t)b * C_ + o) * N_ + p] = val;
      outPad[(((size_t)b * NH_ + h) * N_ + p) * HCP_ + c] = (__bf16)(val * SCALE_);
    } else if (mode == 1) {
      outPad[(((size_t)b * NH_ + h) * N_ + p) * HCP_ + c] = (__bf16)val;
    } else if (mode == 2) {
      outRow[(((size_t)b * NH_ + h) * HC_ + c) * N_ + p] = (__bf16)val;
    } else {
      outF[((size_t)b * C_ + o) * N_ + p] = val;
    }
  }
  // zero the K-dim pad (c = 48..63) once per head; done by the o-tile covering c=32..47
  if ((mode == 0 || mode == 1) && (o0 % HC_) == 32) {
    int h = o0 / HC_;
    #pragma unroll
    for (int r = 0; r < 8; ++r) {
      int cpad = HC_ + r + 8 * g;
      int p = p0 + ln;
      outPad[(((size_t)b * NH_ + h) * N_ + p) * HCP_ + cpad] = (__bf16)0.0f;
    }
  }
}

// ---------------------------------------------------------------------------
// dwconv3x3 + LayerNorm(C) + GELU(exact) + offset einsum + ref points -> grid
// One wave per (group-batch, pixel); lane handles channels {l, l+32, l+64}.
// gridbuf[(bg*n+p)*2+0] = x-coord (normalized), +1 = y-coord
// ---------------------------------------------------------------------------
__global__ void k_offset_grid(const float* __restrict__ q, const float* __restrict__ wdw,
                              const float* __restrict__ bdw, const float* __restrict__ lnw,
                              const float* __restrict__ lnb, const float* __restrict__ woo,
                              float* __restrict__ gridbuf) {
  int p = blockIdx.x, bg = blockIdx.y, lane = threadIdx.x;
  int y = p / W_, xc = p % W_;
  int b = bg >> 2, gidx = bg & 3;
  const float* qg = q + ((size_t)b * C_ + gidx * GC_) * N_;

  float t[3];
  #pragma unroll
  for (int j = 0; j < 3; ++j) {
    int c = lane + 32 * j;
    const float* img = qg + (size_t)c * N_;
    const float* wp  = wdw + c * 9;
    float acc = bdw[c];
    #pragma unroll
    for (int dy = -1; dy <= 1; ++dy)
      #pragma unroll
      for (int dx = -1; dx <= 1; ++dx) {
        int yy = y + dy, xx = xc + dx;
        float v = (yy >= 0 && yy < H_ && xx >= 0 && xx < W_) ? img[yy * W_ + xx] : 0.0f;
        acc += v * wp[(dy + 1) * 3 + (dx + 1)];
      }
    t[j] = acc;
  }
  float s  = t[0] + t[1] + t[2];
  float s2 = t[0]*t[0] + t[1]*t[1] + t[2]*t[2];
  s = wave_sum32(s); s2 = wave_sum32(s2);
  float mu  = s * (1.0f / GC_);
  float var = s2 * (1.0f / GC_) - mu * mu;
  float rs  = rsqrtf(var + 1e-5f);

  float o0 = 0.0f, o1 = 0.0f;
  #pragma unroll
  for (int j = 0; j < 3; ++j) {
    int c = lane + 32 * j;
    float xn = (t[j] - mu) * rs * lnw[c] + lnb[c];
    float ge = 0.5f * xn * (1.0f + erff(xn * 0.70710678118654752f));
    o0 += ge * woo[c];
    o1 += ge * woo[GC_ + c];
  }
  o0 = wave_sum32(o0); o1 = wave_sum32(o1);
  if (lane == 0) {
    float ry = ((float)y  + 0.5f) * (2.0f / (H_ - 1)) - 1.0f;
    float rx = ((float)xc + 0.5f) * (2.0f / (W_ - 1)) - 1.0f;
    gridbuf[((size_t)bg * N_ + p) * 2 + 0] = o1 + rx;  // grid x
    gridbuf[((size_t)bg * N_ + p) * 2 + 1] = o0 + ry;  // grid y
  }
}

// ---------------------------------------------------------------------------
// Bilinear grid sample (zero-pad) -> xsb (B, n, C) bf16, transposed for GEMMs
// ---------------------------------------------------------------------------
__global__ void k_sample(const float* __restrict__ x, const float* __restrict__ gridbuf,
                         __bf16* __restrict__ xsb) {
  int p = blockIdx.x, bg = blockIdx.y, c = threadIdx.x;
  float gxn = gridbuf[((size_t)bg * N_ + p) * 2 + 0];
  float gyn = gridbuf[((size_t)bg * N_ + p) * 2 + 1];
  float gx = (gxn + 1.0f) * 0.5f * (float)(W_ - 1);
  float gy = (gyn + 1.0f) * 0.5f * (float)(H_ - 1);
  float x0 = floorf(gx), y0 = floorf(gy);
  float x1 = x0 + 1.0f,  y1 = y0 + 1.0f;
  int b = bg >> 2, gidx = bg & 3;
  const float* img = x + ((size_t)b * C_ + gidx * GC_ + c) * N_;

  auto fetch = [&](float xi, float yi) -> float {
    int xv = (int)fminf(fmaxf(xi, 0.0f), (float)(W_ - 1));
    int yv = (int)fminf(fmaxf(yi, 0.0f), (float)(H_ - 1));
    float m = (xi >= 0.0f && xi <= (float)(W_ - 1) &&
               yi >= 0.0f && yi <= (float)(H_ - 1)) ? 1.0f : 0.0f;
    return img[yv * W_ + xv] * m;
  };
  float v00 = fetch(x0, y0), v01 = fetch(x1, y0);
  float v10 = fetch(x0, y1), v11 = fetch(x1, y1);
  float w00 = (x1 - gx) * (y1 - gy), w01 = (gx - x0) * (y1 - gy);
  float w10 = (x1 - gx) * (gy - y0), w11 = (gx - x0) * (gy - y0);
  float val = v00 * w00 + v01 * w01 + v10 * w10 + v11 * w11;
  xsb[((size_t)b * N_ + p) * C_ + gidx * GC_ + c] = (__bf16)val;
}

// ---------------------------------------------------------------------------
// lepe = depthwise conv3x3(q) + b_rpe, fp32 (B, C, n)
// ---------------------------------------------------------------------------
__global__ void k_lepe(const float* __restrict__ q, const float* __restrict__ wrpe,
                       const float* __restrict__ brpe, float* __restrict__ lepe) {
  int tid = blockIdx.x * 256 + threadIdx.x;
  if (tid >= B_ * C_ * N_) return;
  int p = tid % N_;
  int c = (tid / N_) % C_;
  int b = tid / (N_ * C_);
  int y = p / W_, xc = p % W_;
  const float* img = q + ((size_t)b * C_ + c) * N_;
  const float* wp  = wrpe + c * 9;
  float acc = brpe[c];
  #pragma unroll
  for (int dy = -1; dy <= 1; ++dy)
    #pragma unroll
    for (int dx = -1; dx <= 1; ++dx) {
      int yy = y + dy, xx = xc + dx;
      float v = (yy >= 0 && yy < H_ && xx >= 0 && xx < W_) ? img[yy * W_ + xx] : 0.0f;
      acc += v * wp[(dy + 1) * 3 + (dx + 1)];
    }
  lepe[tid] = acc;
}

// ---------------------------------------------------------------------------
// Flash attention: per wave one (head-batch, 16-query tile), stream KV in
// 32-wide chunks. QK^T: 4 WMMA; PV: 3 WMMA per chunk (c tiles 0/16/32).
// P tile goes D-layout -> A-layout through a 1KB/wave LDS round-trip.
// ---------------------------------------------------------------------------
__global__ void k_flash_attn(const __bf16* __restrict__ qp, const __bf16* __restrict__ kp,
                             const __bf16* __restrict__ vr, const float* __restrict__ lepe,
                             __bf16* __restrict__ outb) {
  __shared__ __bf16 plds[4][16 * 32];
  int lane = threadIdx.x & 31, w = threadIdx.x >> 5;
  int ln = lane & 15, g = lane >> 4;
  int m0 = (blockIdx.x * 4 + w) * 16;
  int bh = blockIdx.y;
  int b = bh >> 3, h = bh & 7;

  const __bf16* qrow = qp + ((size_t)bh * N_ + m0 + ln) * HCP_;
  v16bf QA0 = load_frag(qrow + 16 * g,      qrow + 16 * g + 8);   // c = 0..31
  v16bf QA1 = load_frag(qrow + 32 + 16 * g, qrow + 40 + 16 * g);   // c = 32..63 (pad=0)

  float mx[8], sm[8];
  v8f O0 = {}, O1 = {}, O2 = {};
  #pragma unroll
  for (int r = 0; r < 8; ++r) { mx[r] = -3.0e38f; sm[r] = 0.0f; }

  for (int n0 = 0; n0 < N_; n0 += 32) {
    v8f S0 = {}, S1 = {};
    const __bf16* k0 = kp + ((size_t)bh * N_ + n0 + ln) * HCP_;
    const __bf16* k1 = kp + ((size_t)bh * N_ + n0 + 16 + ln) * HCP_;
    v16bf Bk;
    Bk = load_frag(k0 + 16 * g,      k0 + 16 * g + 8);  S0 = WMMA_BF16(QA0, Bk, S0);
    Bk = load_frag(k0 + 32 + 16 * g, k0 + 40 + 16 * g); S0 = WMMA_BF16(QA1, Bk, S0);
    Bk = load_frag(k1 + 16 * g,      k1 + 16 * g + 8);  S1 = WMMA_BF16(QA0, Bk, S1);
    Bk = load_frag(k1 + 32 + 16 * g, k1 + 40 + 16 * g); S1 = WMMA_BF16(QA1, Bk, S1);

    #pragma unroll
    for (int r = 0; r < 8; ++r) {
      float m = fmaxf(S0[r], S1[r]);
      #pragma unroll
      for (int off = 1; off < 16; off <<= 1) m = fmaxf(m, __shfl_xor(m, off, 32));
      float nm   = fmaxf(mx[r], m);
      float corr = __expf(mx[r] - nm);
      mx[r] = nm;
      float p0v = __expf(S0[r] - nm);
      float p1v = __expf(S1[r] - nm);
      float ps = p0v + p1v;
      #pragma unroll
      for (int off = 1; off < 16; off <<= 1) ps += __shfl_xor(ps, off, 32);
      sm[r] = sm[r] * corr + ps;
      O0[r] *= corr; O1[r] *= corr; O2[r] *= corr;
      plds[w][(r + 8 * g) * 32 + ln]      = (__bf16)p0v;   // row m, col n0..15
      plds[w][(r + 8 * g) * 32 + 16 + ln] = (__bf16)p1v;   // row m, col n16..31
    }
    // D-layout -> A-layout via LDS (same-wave dependency; HW keeps DS in order)
    v16bf PA = load_frag(&plds[w][ln * 32 + 16 * g], &plds[w][ln * 32 + 16 * g + 8]);

    const __bf16* vb = vr + (size_t)bh * HC_ * N_ + n0 + 16 * g;
    v16bf Bv;
    Bv = load_frag(vb + (size_t)(0  + ln) * N_, vb + (size_t)(0  + ln) * N_ + 8); O0 = WMMA_BF16(PA, Bv, O0);
    Bv = load_frag(vb + (size_t)(16 + ln) * N_, vb + (size_t)(16 + ln) * N_ + 8); O1 = WMMA_BF16(PA, Bv, O1);
    Bv = load_frag(vb + (size_t)(32 + ln) * N_, vb + (size_t)(32 + ln) * N_ + 8); O2 = WMMA_BF16(PA, Bv, O2);
  }

  const float* lrow = lepe + ((size_t)b * C_ + h * HC_) * N_;
  #pragma unroll
  for (int r = 0; r < 8; ++r) {
    float rinv = 1.0f / sm[r];
    int m = m0 + r + 8 * g;
    float v0 = O0[r] * rinv + lrow[(size_t)(0  + ln) * N_ + m];
    float v1 = O1[r] * rinv + lrow[(size_t)(16 + ln) * N_ + m];
    float v2 = O2[r] * rinv + lrow[(size_t)(32 + ln) * N_ + m];
    __bf16* orow = outb + ((size_t)b * N_ + m) * C_ + h * HC_;
    orow[0  + ln] = (__bf16)v0;
    orow[16 + ln] = (__bf16)v1;
    orow[32 + ln] = (__bf16)v2;
  }
}

// ---------------------------------------------------------------------------
extern "C" void kernel_launch(void* const* d_in, const int* in_sizes, int n_in,
                              void* d_out, int out_size, void* d_ws, size_t ws_size,
                              hipStream_t stream) {
  const float* x        = (const float*)d_in[0];
  const float* wq       = (const float*)d_in[1];
  const float* bq       = (const float*)d_in[2];
  const float* w_off_dw = (const float*)d_in[3];
  const float* b_off_dw = (const float*)d_in[4];
  const float* ln_w     = (const float*)d_in[5];
  const float* ln_b     = (const float*)d_in[6];
  const float* w_off_out= (const float*)d_in[7];
  const float* wk       = (const float*)d_in[8];
  const float* bk       = (const float*)d_in[9];
  const float* wv       = (const float*)d_in[10];
  const float* bv       = (const float*)d_in[11];
  const float* wo       = (const float*)d_in[12];
  const float* bo       = (const float*)d_in[13];
  const float* w_rpe    = (const float*)d_in[14];
  const float* b_rpe    = (const float*)d_in[15];
  float* out = (float*)d_out;

  char* ws = (char*)d_ws;
  size_t cur = 0;
  auto alloc = [&](size_t bytes) -> void* {
    void* p = ws + cur;
    cur = (cur + bytes + 255) & ~(size_t)255;
    return p;
  };

  const size_t WN = (size_t)C_ * C_;                 // 147456
  __bf16* wqb   = (__bf16*)alloc(WN * 2);
  __bf16* wkb   = (__bf16*)alloc(WN * 2);
  __bf16* wvb   = (__bf16*)alloc(WN * 2);
  __bf16* wob   = (__bf16*)alloc(WN * 2);
  __bf16* xb    = (__bf16*)alloc((size_t)B_ * N_ * C_ * 2);
  float*  qf32  = (float*) alloc((size_t)B_ * C_ * N_ * 4);
  __bf16* qpb   = (__bf16*)alloc((size_t)B_ * NH_ * N_ * HCP_ * 2);
  __bf16* kpb   = (__bf16*)alloc((size_t)B_ * NH_ * N_ * HCP_ * 2);
  __bf16* vrow  = (__bf16*)alloc((size_t)B_ * NH_ * HC_ * N_ * 2);
  float*  gridb = (float*) alloc((size_t)B_ * NG_ * N_ * 2 * 4);
  __bf16* xsb   = (__bf16*)alloc((size_t)B_ * N_ * C_ * 2);
  float*  lepe  = (float*) alloc((size_t)B_ * C_ * N_ * 4);
  __bf16* outb  = (__bf16*)alloc((size_t)B_ * N_ * C_ * 2);

  // weights -> bf16
  k_cvt_bf16<<<(int)((WN + 255) / 256), 256, 0, stream>>>(wq, wqb, (int)WN);
  k_cvt_bf16<<<(int)((WN + 255) / 256), 256, 0, stream>>>(wk, wkb, (int)WN);
  k_cvt_bf16<<<(int)((WN + 255) / 256), 256, 0, stream>>>(wv, wvb, (int)WN);
  k_cvt_bf16<<<(int)((WN + 255) / 256), 256, 0, stream>>>(wo, wob, (int)WN);

  // x -> (B, n, C) bf16
  k_transpose_x<<<dim3(N_ / 32, C_ / 32, B_), dim3(32, 8), 0, stream>>>(x, xb);

  // q = wq @ x + bq   (fp32 + scaled/padded bf16)
  k_gemm384<<<dim3(25, 24, 8), 128, 0, stream>>>(wqb, xb, bq, 0, qf32, qpb, nullptr);

  // offset branch -> sampling grid
  k_offset_grid<<<dim3(N_, B_ * NG_), 32, 0, stream>>>(qf32, w_off_dw, b_off_dw,
                                                       ln_w, ln_b, w_off_out, gridb);
  // bilinear sample -> xsb (B, n, C) bf16
  k_sample<<<dim3(N_, B_ * NG_), GC_, 0, stream>>>(x, gridb, xsb);

  // k, v projections
  k_gemm384<<<dim3(25, 24, 8), 128, 0, stream>>>(wkb, xsb, bk, 1, nullptr, kpb, nullptr);
  k_gemm384<<<dim3(25, 24, 8), 128, 0, stream>>>(wvb, xsb, bv, 2, nullptr, nullptr, vrow);

  // lepe
  k_lepe<<<(B_ * C_ * N_ + 255) / 256, 256, 0, stream>>>(qf32, w_rpe, b_rpe, lepe);

  // streaming softmax attention + lepe add -> outb (B, n, C) bf16
  k_flash_attn<<<dim3(25, B_ * NH_), 128, 0, stream>>>(qpb, kpb, vrow, lepe, outb);

  // final projection -> d_out fp32
  k_gemm384<<<dim3(25, 24, 8), 128, 0, stream>>>(wob, outb, bo, 3, out, nullptr, nullptr);
}
